// LocalMultiheadAttention_73615739453806
// MI455X (gfx1250) — matrix operand order repaired and checked by reading, not compiled, not run on hardware
//
#include <hip/hip_runtime.h>
#include <stdint.h>

#define D_EMBED 1024
#define NHEAD   16
#define DHEAD   64
#define D3      3072
#define BB      4
#define TT      2048
#define MTOT    (BB * TT)     // 8192 tokens
#define NTILES  (TT / 16)     // 128 query tiles per (b,h)

typedef __attribute__((ext_vector_type(16))) __bf16 v16bf;
typedef __attribute__((ext_vector_type(8)))  float  v8f;
typedef __attribute__((ext_vector_type(4)))  unsigned int u32x4;
typedef __attribute__((ext_vector_type(8)))  int i32x8;
typedef __attribute__((ext_vector_type(4)))  int i32x4;

#if defined(__HIP_DEVICE_COMPILE__) && __has_builtin(__builtin_amdgcn_tensor_load_to_lds) && __has_builtin(__builtin_amdgcn_s_wait_tensorcnt)
#define HAVE_TDM 1
#else
#define HAVE_TDM 0
#endif

struct B32x8 { u32x4 lo, hi; };

static __device__ __forceinline__ v16bf frag_cast(u32x4 lo, u32x4 hi) {
  B32x8 t; t.lo = lo; t.hi = hi;
  return __builtin_bit_cast(v16bf, t);
}

static __device__ __forceinline__ v8f wmma_bf16(v16bf a, v16bf b, v8f c) {
  // D = A(16x32 bf16) * B(32x16 bf16) + C(16x16 f32)
  return __builtin_amdgcn_wmma_f32_16x16x32_bf16(false, a, false, b, (short)0, c,
                                                 false, false);
}

// round-to-nearest-even f32 -> bf16 bits
static __device__ __forceinline__ unsigned short f2bf(float f) {
  unsigned int u = __builtin_bit_cast(unsigned int, f);
  u += 0x7FFFu + ((u >> 16) & 1u);
  return (unsigned short)(u >> 16);
}

#if HAVE_TDM
// TDM: DMA a 64-row x 32-col bf16 tile (row stride K elems) from global into LDS,
// padding LDS by 4 DW after every 16 DW stored -> LDS row stride 40 ushorts.
// D# layout per CDNA5 ISA (group0 128b, group1 256b).
static __device__ __forceinline__ void tdm_load_w_tile(const unsigned short* gsrc,
                                                       unsigned int lds_off, int K) {
  union { unsigned int w[4]; u32x4 v; } g0;
  unsigned long long ga = (unsigned long long)(uintptr_t)gsrc;
  g0.w[0] = 1u;                                  // count=1, is_restore=0, no gather
  g0.w[1] = lds_off;                             // lds_addr [63:32]
  g0.w[2] = (unsigned int)ga;                    // global_addr [95:64]
  g0.w[3] = (unsigned int)((ga >> 32) & 0x01FFFFFFu) | (2u << 30);  // addr hi + type=2

  union { int w[8]; i32x8 v; } g1;
  // data_size=1 (2B) <<16 | pad_enable<<20 | pad_interval=3 (16DW) <<22 | pad_amount=3 (4DW) <<25
  g1.w[0] = (1 << 16) | (1 << 20) | (3 << 22) | (3 << 25);
  g1.w[1] = (32 << 16);          // tensor_dim0[15:0]=32 at bits 63:48
  g1.w[2] = (64 << 16);          // tensor_dim0 hi=0 | tensor_dim1[15:0]=64 at 95:80
  g1.w[3] = (32 << 16);          // tensor_dim1 hi=0 | tile_dim0=32 at 127:112
  g1.w[4] = 64;                  // tile_dim1=64 at 143:128, tile_dim2=0
  g1.w[5] = K;                   // tensor_dim0_stride[31:0]
  g1.w[6] = 0;                   // stride hi, tensor_dim1_stride lo (unused, 2D)
  g1.w[7] = 0;

  i32x4 gz  = {0, 0, 0, 0};
#if __clang_major__ >= 23
  i32x8 gz8 = {0, 0, 0, 0, 0, 0, 0, 0};
  __builtin_amdgcn_tensor_load_to_lds(g0.v, g1.v, gz, gz, gz8, 0);
#else
  __builtin_amdgcn_tensor_load_to_lds(g0.v, g1.v, gz, gz, 0);
#endif
}
#endif

__global__ __launch_bounds__(256) void cvt_f32_to_bf16(
    const float* __restrict__ in, unsigned short* __restrict__ out, int n) {
  int i = blockIdx.x * blockDim.x + threadIdx.x;
  int stride = gridDim.x * blockDim.x;
  for (; i < n; i += stride) out[i] = f2bf(in[i]);
}

// C[M,N] = A[M,K](bf16) * W[N,K]^T(bf16) + bias ; OUT_BF16 selects bf16/f32 output.
// Block = 256 threads (8 waves). Block tile M=128, N=64. Wave tile 16x64.
// W tile staged to LDS via the Tensor Data Mover; A tile via VMEM+DS with prefetch.
template <int OUT_BF16>
__global__ __launch_bounds__(256) void gemm_bf16_kern(
    const unsigned short* __restrict__ A, const unsigned short* __restrict__ W,
    const float* __restrict__ bias, float* __restrict__ Cf,
    unsigned short* __restrict__ Cb, int M, int N, int K) {
  __shared__ unsigned short As[128 * 40];  // stride 40 (80B) to spread banks
  __shared__ unsigned short Ws[64 * 40];

  const int tid = threadIdx.x;
  const int wave = tid >> 5, lane = tid & 31;
  const int l16 = lane & 15, kh = lane >> 4;
  const int m0b = blockIdx.y * 128;
  const int n0 = blockIdx.x * 64;
  const int mw = m0b + wave * 16;

  v8f acc[4];
#pragma unroll
  for (int i = 0; i < 4; ++i) acc[i] = {0.f, 0.f, 0.f, 0.f, 0.f, 0.f, 0.f, 0.f};

  const int ar = tid >> 1, ah = tid & 1;

#if HAVE_TDM
  const unsigned int ws_lds = (unsigned int)(uintptr_t)(void*)&Ws[0];
#endif

  for (int kk = 0; kk < K; kk += 32) {
    {  // stage A tile: 128 rows x 32 cols, 32B per thread
      const u32x4* src = (const u32x4*)(A + (size_t)(m0b + ar) * K + kk + ah * 16);
      u32x4 v0 = src[0], v1 = src[1];
      if (kk + 32 < K)  // pull next A chunk toward the WGP (global_prefetch_b8)
        __builtin_prefetch(A + (size_t)(m0b + ar) * K + kk + 32 + ah * 16, 0, 1);
      u32x4* dst = (u32x4*)(As + ar * 40 + ah * 16);
      dst[0] = v0; dst[1] = v1;
    }
#if HAVE_TDM
    if (wave == 0) {
      tdm_load_w_tile(W + (size_t)n0 * K + kk, ws_lds, K);
    }
    __builtin_amdgcn_s_wait_tensorcnt(0);  // no-op for waves with TENSORcnt==0
#else
    if (tid < 128) {  // stage W tile: 64 rows x 32 cols
      const u32x4* src = (const u32x4*)(W + (size_t)(n0 + ar) * K + kk + ah * 16);
      u32x4 v0 = src[0], v1 = src[1];
      u32x4* dst = (u32x4*)(Ws + ar * 40 + ah * 16);
      dst[0] = v0; dst[1] = v1;
    }
#endif
    __syncthreads();

    // A fragment: lane row = l16, K chunks [kh*8, +8) and [16+kh*8, +8)
    const int am = wave * 16 + l16;
    u32x4 a0 = *(const u32x4*)(As + am * 40 + kh * 8);
    u32x4 a1 = *(const u32x4*)(As + am * 40 + 16 + kh * 8);
    v16bf afrag = frag_cast(a0, a1);

#pragma unroll
    for (int nt = 0; nt < 4; ++nt) {
      // B fragment: lane col = nt*16+l16, K chunks [kh*8, +8) and [16+kh*8, +8)
      const unsigned short* bp = Ws + (nt * 16 + l16) * 40 + kh * 8;
      u32x4 b0 = *(const u32x4*)(bp);
      u32x4 b1 = *(const u32x4*)(bp + 16);
      acc[nt] = wmma_bf16(afrag, frag_cast(b0, b1), acc[nt]);
    }
    __syncthreads();
  }

#pragma unroll
  for (int nt = 0; nt < 4; ++nt) {
    int n = n0 + nt * 16 + l16;
    float bv = bias[n];
#pragma unroll
    for (int r = 0; r < 8; ++r) {
      int m = mw + r + 8 * kh;  // C layout: lanes 16-31 hold M = r+8
      float v = acc[nt][r] + bv;
      if (OUT_BF16) Cb[(size_t)m * N + n] = f2bf(v);
      else          Cf[(size_t)m * N + n] = v;
    }
  }
}

// One wave per (b, h, 16-query tile). Window +-32 -> 5 key tiles of 16.
__global__ __launch_bounds__(128) void attn_kern(
    const unsigned short* __restrict__ qkv, unsigned short* __restrict__ ctx) {
  __shared__ unsigned short sm[4][16 * 96 + 64 * 96];  // per-wave: P[16x96], VT[64x96]

  const int tid = threadIdx.x;
  const int wave = tid >> 5, lane = tid & 31;
  const int l16 = lane & 15, kh = lane >> 4;
  const int w = blockIdx.x * 4 + wave;
  const int qt = w & (NTILES - 1);
  const int bh = w >> 7;
  const int h = bh & (NHEAD - 1);
  const int b = bh >> 4;
  const int i0 = qt * 16;

  unsigned short* P = sm[wave];
  unsigned short* VT = sm[wave] + 16 * 96;

  // ---- Q fragments (A layout), directly from global: two 16B chunks each ----
  size_t qbase = ((size_t)(b * TT + i0 + l16)) * D3 + h * DHEAD;
  v16bf aq0 = frag_cast(*(const u32x4*)(qkv + qbase + kh * 8),
                        *(const u32x4*)(qkv + qbase + 16 + kh * 8));
  v16bf aq1 = frag_cast(*(const u32x4*)(qkv + qbase + 32 + kh * 8),
                        *(const u32x4*)(qkv + qbase + 48 + kh * 8));

  // ---- scores: S[t] = Q * K_tile^T  (2 WMMAs per key tile over d=64) ----
  float s[5][8];
  const v8f vzero = {0.f, 0.f, 0.f, 0.f, 0.f, 0.f, 0.f, 0.f};
#pragma unroll
  for (int t = 0; t < 5; ++t) {
    int jt = qt - 2 + t;
    if (jt >= 0 && jt < NTILES) {
      int j0 = jt * 16;
      size_t kb = ((size_t)(b * TT + j0 + l16)) * D3 + D_EMBED + h * DHEAD;
      v16bf bk0 = frag_cast(*(const u32x4*)(qkv + kb + kh * 8),
                            *(const u32x4*)(qkv + kb + 16 + kh * 8));
      v16bf bk1 = frag_cast(*(const u32x4*)(qkv + kb + 32 + kh * 8),
                            *(const u32x4*)(qkv + kb + 48 + kh * 8));
      v8f c = wmma_bf16(aq0, bk0, vzero);
      c = wmma_bf16(aq1, bk1, c);
#pragma unroll
      for (int r = 0; r < 8; ++r) {
        int i = i0 + r + 8 * kh;
        int j = j0 + l16;
        int dij = i - j; if (dij < 0) dij = -dij;
        s[t][r] = (dij <= 32) ? c[r] * 0.125f : -__builtin_inff();
      }
    } else {
#pragma unroll
      for (int r = 0; r < 8; ++r) s[t][r] = -__builtin_inff();
    }
  }

  // ---- softmax over 80 keys; row reductions within 16-lane half ----
#pragma unroll
  for (int r = 0; r < 8; ++r) {
    float mx = s[0][r];
#pragma unroll
    for (int t = 1; t < 5; ++t) mx = fmaxf(mx, s[t][r]);
#pragma unroll
    for (int d = 1; d <= 8; d <<= 1) mx = fmaxf(mx, __shfl_xor(mx, d, 32));
    float e[5];
    float sum = 0.0f;
#pragma unroll
    for (int t = 0; t < 5; ++t) { e[t] = __expf(s[t][r] - mx); sum += e[t]; }
#pragma unroll
    for (int d = 1; d <= 8; d <<= 1) sum += __shfl_xor(sum, d, 32);
    float inv = __builtin_amdgcn_rcpf(sum);
    int m = r + 8 * kh;
#pragma unroll
    for (int t = 0; t < 5; ++t) P[m * 96 + t * 16 + l16] = f2bf(e[t] * inv);
    P[m * 96 + 80 + l16] = 0;  // zero pad cols 80..95
  }

  // ---- stage V transposed: VT[d][j_local], zero-fill invalid/pad tiles ----
  const u32x4 z4 = {0u, 0u, 0u, 0u};
#pragma unroll
  for (int t = 0; t < 6; ++t) {
    int jt = qt - 2 + t;
    if (t < 5 && jt >= 0 && jt < NTILES) {
      int jl = t * 16 + (lane >> 1);
      int j = jt * 16 + (lane >> 1);
      int dbase = (lane & 1) * 32;
      size_t vb = ((size_t)(b * TT + j)) * D3 + 2 * D_EMBED + h * DHEAD + dbase;
      union { u32x4 q[4]; unsigned short u[32]; } uv;
      const u32x4* src = (const u32x4*)(qkv + vb);
      uv.q[0] = src[0]; uv.q[1] = src[1]; uv.q[2] = src[2]; uv.q[3] = src[3];
#pragma unroll
      for (int e2 = 0; e2 < 32; ++e2) VT[(dbase + e2) * 96 + jl] = uv.u[e2];
    } else {
#pragma unroll
      for (int rr = 0; rr < 2; ++rr) {
        int d = lane + rr * 32;
        u32x4* dst = (u32x4*)(VT + d * 96 + t * 16);
        dst[0] = z4; dst[1] = z4;
      }
    }
  }

  asm volatile("" ::: "memory");  // keep LDS store->load order (per-wave in-order)

  // ---- O = P(16x96) * V(96x64): 3 k-steps x 4 n-tiles of WMMA ----
  v8f o[4];
#pragma unroll
  for (int i = 0; i < 4; ++i) o[i] = {0.f, 0.f, 0.f, 0.f, 0.f, 0.f, 0.f, 0.f};
#pragma unroll
  for (int ks = 0; ks < 3; ++ks) {
    int kk = ks * 32;
    v16bf ap = frag_cast(*(const u32x4*)(P + l16 * 96 + kk + kh * 8),
                         *(const u32x4*)(P + l16 * 96 + kk + 16 + kh * 8));
#pragma unroll
    for (int nt = 0; nt < 4; ++nt) {
      const unsigned short* bp = VT + (nt * 16 + l16) * 96 + kk + kh * 8;
      v16bf bv = frag_cast(*(const u32x4*)bp, *(const u32x4*)(bp + 16));
      o[nt] = wmma_bf16(ap, bv, o[nt]);
    }
  }

  // ---- write context (bf16) ----
#pragma unroll
  for (int nt = 0; nt < 4; ++nt) {
#pragma unroll
    for (int r = 0; r < 8; ++r) {
      int i = i0 + r + 8 * kh;
      int col = h * DHEAD + nt * 16 + l16;
      ctx[(size_t)(b * TT + i) * D_EMBED + col] = f2bf(o[nt][r]);
    }
  }
}

extern "C" void kernel_launch(void* const* d_in, const int* in_sizes, int n_in,
                              void* d_out, int out_size, void* d_ws, size_t ws_size,
                              hipStream_t stream) {
  const float* x     = (const float*)d_in[0];
  const float* W_qkv = (const float*)d_in[1];
  const float* b_qkv = (const float*)d_in[2];
  const float* W_out = (const float*)d_in[3];
  const float* b_out = (const float*)d_in[4];

  char* ws = (char*)d_ws;
  unsigned short* xb    = (unsigned short*)(ws);              // 16 MB
  unsigned short* wqkvb = (unsigned short*)(ws + 16777216);   // 6 MB
  unsigned short* woutb = (unsigned short*)(ws + 23068672);   // 2 MB
  unsigned short* qkvb  = (unsigned short*)(ws + 25165824);   // 48 MB
  unsigned short* ctxb  = (unsigned short*)(ws + 75497472);   // 16 MB

  cvt_f32_to_bf16<<<4096, 256, 0, stream>>>(x, xb, MTOT * D_EMBED);
  cvt_f32_to_bf16<<<2048, 256, 0, stream>>>(W_qkv, wqkvb, D3 * D_EMBED);
  cvt_f32_to_bf16<<<1024, 256, 0, stream>>>(W_out, woutb, D_EMBED * D_EMBED);

  gemm_bf16_kern<1><<<dim3(D3 / 64, MTOT / 128), 256, 0, stream>>>(
      xb, wqkvb, b_qkv, nullptr, qkvb, MTOT, D3, D_EMBED);

  attn_kern<<<(BB * NHEAD * NTILES) / 4, 128, 0, stream>>>(qkvb, ctxb);

  gemm_bf16_kern<0><<<dim3(D_EMBED / 64, MTOT / 128), 256, 0, stream>>>(
      ctxb, woutb, b_out, (float*)d_out, nullptr, MTOT, D_EMBED, D_EMBED);
}
